// FK_44495861186601
// MI455X (gfx1250) — matrix operand order repaired and checked by reading, not compiled, not run on hardware
//
#include <hip/hip_runtime.h>
#include <hip/hip_bf16.h>
#include <math.h>

typedef __bf16 bf16_t;
typedef __attribute__((ext_vector_type(16))) __bf16 v16bf;
typedef __attribute__((ext_vector_type(8)))  __bf16 v8bf;
typedef __attribute__((ext_vector_type(4)))  __bf16 v4bf;
typedef __attribute__((ext_vector_type(8)))  float  v8f;
typedef __attribute__((ext_vector_type(4)))  float  f32x4;

#define WG   256
#define NKER 11
#define TM 128
#define TN 64
#define TK 32
#define KPAD 8                    // rows of (TK+KPAD)=40 halves = 80 B (16B-aligned)
#define RSTR (TK + KPAD)

#define SHUF16(lo, hi) __builtin_shufflevector((lo), (hi), 0,1,2,3,4,5,6,7,8,9,10,11,12,13,14,15)
#define WMMA_BF16(a, b, c) __builtin_amdgcn_wmma_f32_16x16x32_bf16(false, (a), false, (b), (short)0, (c), false, false)

// ---------------------------------------------------------------------------
// DFT matrix generation: cm[j,k]=cos(2*pi*j*k/n), sm[j,k]=sin(2*pi*j*k/n)
// ---------------------------------------------------------------------------
__global__ __launch_bounds__(WG)
void dft_gen_kernel(float* __restrict__ cm, float* __restrict__ sm, int n)
{
    long long idx = (long long)blockIdx.x * WG + threadIdx.x;
    long long total = (long long)n * n;
    if (idx >= total) return;
    int j = (int)(idx / n);
    int k = (int)(idx % n);
    long long jk = ((long long)j * (long long)k) % n;
    float ang = (float)(6.28318530717958647692 / (double)n * (double)jk);
    cm[idx] = cosf(ang);
    sm[idx] = sinf(ang);
}

// ---------------------------------------------------------------------------
// Embedding gather + sinusoidal positional features.
// blockDim.x == 256 == E, one block per token.  pos written to 1 or 2 outputs.
// ---------------------------------------------------------------------------
__global__ __launch_bounds__(WG)
void embed_pos_kernel(const int* __restrict__ toks, const float* __restrict__ table,
                      float* __restrict__ emb_out, float* __restrict__ pos_out,
                      float* __restrict__ pos_out2, int Lseq)
{
    int row = blockIdx.x;
    int e   = threadIdx.x;
    int tok = toks[row];
    float v = table[(long long)tok * 256 + e];
    if (emb_out) emb_out[(long long)row * 256 + e] = v;
    int p = row % Lseq;
    const float inc = 9.210340371976184f / 127.0f;   // ln(1e4)/(nt-1), nt=128
    float s;
    if (e < 128) s = __sinf((float)p * __expf(-inc * (float)e));
    else         s = __cosf((float)p * __expf(-inc * (float)(e - 128)));
    float o = v + s;
    pos_out[(long long)row * 256 + e] = o;
    if (pos_out2) pos_out2[(long long)row * 256 + e] = o;
}

// ===========================================================================
// GEMM family.  Common facts for every call site:
//   N == 256 (multiple of TN), K in {32,256,1024} (multiple of TK);
//   only M may be ragged (query token-mix M=32) -> branch-free row masking.
// WG tile 128x64, 8 waves = 4(M) x 2(N), wave tile 32x32 per output.
// LDS tiles K-contiguous (B transposed) -> fragments = two ds_load_b128.
// ===========================================================================

// ---- single GEMM with bias:  C = A@B + bias  (FF layers) -------------------
__global__ __launch_bounds__(WG)
void gemm_wmma_kernel(const float* __restrict__ A, const float* __restrict__ B,
                      float* __restrict__ C, int M, int N, int Kd,
                      const float* __restrict__ bias)
{
    __shared__ alignas(16) bf16_t Asm[TM][RSTR];
    __shared__ alignas(16) bf16_t Bsm[TN][RSTR];

    int m0 = blockIdx.y * TM, n0 = blockIdx.x * TN;
    int tid = threadIdx.x, wave = tid >> 5, lane = tid & 31;
    int wm = wave & 3, wn = wave >> 2;
    int l15 = lane & 15;
    int akb = (lane >> 4) << 3, bkb = (lane >> 4) << 3;

    // staging pointers (loop-invariant bases)
    const float* apt[4]; float amk[4]; bf16_t* ast[4];
    #pragma unroll
    for (int j = 0; j < 4; ++j) {
        int i = tid + j * WG, r = i >> 3, c4 = (i & 7) << 2;
        int gr = m0 + r;
        amk[j] = (gr < M) ? 1.0f : 0.0f;
        apt[j] = A + (long long)((gr < M) ? gr : 0) * Kd + c4;
        ast[j] = &Asm[r][c4];
    }
    const float* bpt[2]; bf16_t* bst[2];
    #pragma unroll
    for (int j = 0; j < 2; ++j) {
        int i = tid + j * WG, r = i >> 4, c4 = (i & 15) << 2;
        bpt[j] = B + (long long)r * N + (n0 + c4);
        bst[j] = &Bsm[c4][r];
    }
    // fragment read pointers (loop-invariant)
    const bf16_t* ar0 = &Asm[wm * 32 + l15][0];
    const bf16_t* ar1 = &Asm[wm * 32 + 16 + l15][0];
    const bf16_t* bp0 = &Bsm[wn * 32 + l15][bkb];
    const bf16_t* bp1 = &Bsm[wn * 32 + 16 + l15][bkb];

    v8f c00 = {}, c01 = {}, c10 = {}, c11 = {};

    for (int k0 = 0; k0 < Kd; k0 += TK) {
        f32x4 av[4], bv[2];
        #pragma unroll
        for (int j = 0; j < 4; ++j) { av[j] = *(const f32x4*)apt[j]; apt[j] += TK; }
        #pragma unroll
        for (int j = 0; j < 2; ++j) { bv[j] = *(const f32x4*)bpt[j]; bpt[j] += (long long)TK * N; }
        if (k0 + TK < Kd) __builtin_prefetch(apt[0], 0, 1);
        __syncthreads();                       // prev iter's frag reads done
        #pragma unroll
        for (int j = 0; j < 4; ++j) {
            v4bf h;
            h[0] = (bf16_t)(av[j][0] * amk[j]); h[1] = (bf16_t)(av[j][1] * amk[j]);
            h[2] = (bf16_t)(av[j][2] * amk[j]); h[3] = (bf16_t)(av[j][3] * amk[j]);
            *(v4bf*)ast[j] = h;
        }
        #pragma unroll
        for (int j = 0; j < 2; ++j) {
            bst[j][0 * RSTR] = (bf16_t)bv[j][0]; bst[j][1 * RSTR] = (bf16_t)bv[j][1];
            bst[j][2 * RSTR] = (bf16_t)bv[j][2]; bst[j][3 * RSTR] = (bf16_t)bv[j][3];
        }
        __syncthreads();

        v16bf a0 = SHUF16(*(const v8bf*)(ar0 + akb), *(const v8bf*)(ar0 + 16 + akb));
        v16bf a1 = SHUF16(*(const v8bf*)(ar1 + akb), *(const v8bf*)(ar1 + 16 + akb));
        v16bf b0 = SHUF16(*(const v8bf*)(bp0), *(const v8bf*)(bp0 + 16));
        v16bf b1 = SHUF16(*(const v8bf*)(bp1), *(const v8bf*)(bp1 + 16));
        c00 = WMMA_BF16(a0, b0, c00);
        c01 = WMMA_BF16(a0, b1, c01);
        c10 = WMMA_BF16(a1, b0, c10);
        c11 = WMMA_BF16(a1, b1, c11);
    }

    int rb = (lane >> 4) << 3;
    auto store_acc = [&](v8f& acc, int roff, int coff) {
        int col = n0 + wn * 32 + coff + l15;
        float bv = bias ? bias[col] : 0.0f;
        #pragma unroll
        for (int r = 0; r < 8; ++r) {
            int row = m0 + wm * 32 + roff + rb + r;
            if (row < M) C[(long long)row * N + col] = acc[r] + bv;
        }
    };
    store_acc(c00, 0, 0);
    store_acc(c01, 0, 16);
    store_acc(c10, 16, 0);
    store_acc(c11, 16, 16);
}

// ---- dual-B GEMM:  C1 = A@B1, C2 = A@B2  (E-axis DFT: cos & sin at once) ---
__global__ __launch_bounds__(WG)
void gemm2b_wmma_kernel(const float* __restrict__ A,
                        const float* __restrict__ B1, const float* __restrict__ B2,
                        float* __restrict__ C1, float* __restrict__ C2,
                        int M, int N, int Kd)
{
    __shared__ alignas(16) bf16_t Asm[TM][RSTR];
    __shared__ alignas(16) bf16_t B1sm[TN][RSTR];
    __shared__ alignas(16) bf16_t B2sm[TN][RSTR];

    int m0 = blockIdx.y * TM, n0 = blockIdx.x * TN;
    int tid = threadIdx.x, wave = tid >> 5, lane = tid & 31;
    int wm = wave & 3, wn = wave >> 2;
    int l15 = lane & 15;
    int akb = (lane >> 4) << 3, bkb = (lane >> 4) << 3;

    const float* apt[4]; float amk[4]; bf16_t* ast[4];
    #pragma unroll
    for (int j = 0; j < 4; ++j) {
        int i = tid + j * WG, r = i >> 3, c4 = (i & 7) << 2;
        int gr = m0 + r;
        amk[j] = (gr < M) ? 1.0f : 0.0f;
        apt[j] = A + (long long)((gr < M) ? gr : 0) * Kd + c4;
        ast[j] = &Asm[r][c4];
    }
    const float* b1pt[2]; const float* b2pt[2]; bf16_t* b1st[2]; bf16_t* b2st[2];
    #pragma unroll
    for (int j = 0; j < 2; ++j) {
        int i = tid + j * WG, r = i >> 4, c4 = (i & 15) << 2;
        b1pt[j] = B1 + (long long)r * N + (n0 + c4);
        b2pt[j] = B2 + (long long)r * N + (n0 + c4);
        b1st[j] = &B1sm[c4][r];
        b2st[j] = &B2sm[c4][r];
    }
    const bf16_t* ar0 = &Asm[wm * 32 + l15][0];
    const bf16_t* ar1 = &Asm[wm * 32 + 16 + l15][0];
    const bf16_t* p10 = &B1sm[wn * 32 + l15][bkb];
    const bf16_t* p11 = &B1sm[wn * 32 + 16 + l15][bkb];
    const bf16_t* p20 = &B2sm[wn * 32 + l15][bkb];
    const bf16_t* p21 = &B2sm[wn * 32 + 16 + l15][bkb];

    v8f x00 = {}, x01 = {}, x10 = {}, x11 = {};
    v8f y00 = {}, y01 = {}, y10 = {}, y11 = {};

    for (int k0 = 0; k0 < Kd; k0 += TK) {
        f32x4 av[4], b1v[2], b2v[2];
        #pragma unroll
        for (int j = 0; j < 4; ++j) { av[j] = *(const f32x4*)apt[j]; apt[j] += TK; }
        #pragma unroll
        for (int j = 0; j < 2; ++j) {
            b1v[j] = *(const f32x4*)b1pt[j]; b1pt[j] += (long long)TK * N;
            b2v[j] = *(const f32x4*)b2pt[j]; b2pt[j] += (long long)TK * N;
        }
        if (k0 + TK < Kd) __builtin_prefetch(apt[0], 0, 1);
        __syncthreads();
        #pragma unroll
        for (int j = 0; j < 4; ++j) {
            v4bf h;
            h[0] = (bf16_t)(av[j][0] * amk[j]); h[1] = (bf16_t)(av[j][1] * amk[j]);
            h[2] = (bf16_t)(av[j][2] * amk[j]); h[3] = (bf16_t)(av[j][3] * amk[j]);
            *(v4bf*)ast[j] = h;
        }
        #pragma unroll
        for (int j = 0; j < 2; ++j) {
            b1st[j][0 * RSTR] = (bf16_t)b1v[j][0]; b1st[j][1 * RSTR] = (bf16_t)b1v[j][1];
            b1st[j][2 * RSTR] = (bf16_t)b1v[j][2]; b1st[j][3 * RSTR] = (bf16_t)b1v[j][3];
            b2st[j][0 * RSTR] = (bf16_t)b2v[j][0]; b2st[j][1 * RSTR] = (bf16_t)b2v[j][1];
            b2st[j][2 * RSTR] = (bf16_t)b2v[j][2]; b2st[j][3 * RSTR] = (bf16_t)b2v[j][3];
        }
        __syncthreads();

        v16bf a0 = SHUF16(*(const v8bf*)(ar0 + akb), *(const v8bf*)(ar0 + 16 + akb));
        v16bf a1 = SHUF16(*(const v8bf*)(ar1 + akb), *(const v8bf*)(ar1 + 16 + akb));
        v16bf u0 = SHUF16(*(const v8bf*)(p10), *(const v8bf*)(p10 + 16));
        v16bf u1 = SHUF16(*(const v8bf*)(p11), *(const v8bf*)(p11 + 16));
        v16bf w0 = SHUF16(*(const v8bf*)(p20), *(const v8bf*)(p20 + 16));
        v16bf w1 = SHUF16(*(const v8bf*)(p21), *(const v8bf*)(p21 + 16));
        x00 = WMMA_BF16(a0, u0, x00); x01 = WMMA_BF16(a0, u1, x01);
        x10 = WMMA_BF16(a1, u0, x10); x11 = WMMA_BF16(a1, u1, x11);
        y00 = WMMA_BF16(a0, w0, y00); y01 = WMMA_BF16(a0, w1, y01);
        y10 = WMMA_BF16(a1, w0, y10); y11 = WMMA_BF16(a1, w1, y11);
    }

    int rb = (lane >> 4) << 3;
    auto store_acc = [&](float* __restrict__ C, v8f& acc, int roff, int coff) {
        int col = n0 + wn * 32 + coff + l15;
        #pragma unroll
        for (int r = 0; r < 8; ++r) {
            int row = m0 + wm * 32 + roff + rb + r;
            if (row < M) C[(long long)row * N + col] = acc[r];
        }
    };
    store_acc(C1, x00, 0, 0);  store_acc(C1, x01, 0, 16);
    store_acc(C1, x10, 16, 0); store_acc(C1, x11, 16, 16);
    store_acc(C2, y00, 0, 0);  store_acc(C2, y01, 0, 16);
    store_acc(C2, y10, 16, 0); store_acc(C2, y11, 16, 16);
}

// ---- dual-A GEMM (batched):  C = A1@B1 + sgn*(A2@B2)  (token-axis combine) -
// sgn folded into A2's bf16 conversion.  A1/A2 shared across batch.
__global__ __launch_bounds__(WG)
void gemm2a_wmma_kernel(const float* __restrict__ A1, const float* __restrict__ A2,
                        const float* __restrict__ B1, const float* __restrict__ B2,
                        float* __restrict__ C,
                        int M, int N, int Kd,
                        long long sB, long long sC, float sgn)
{
    __shared__ alignas(16) bf16_t A1sm[TM][RSTR];
    __shared__ alignas(16) bf16_t A2sm[TM][RSTR];
    __shared__ alignas(16) bf16_t B1sm[TN][RSTR];
    __shared__ alignas(16) bf16_t B2sm[TN][RSTR];

    const float* B1b = B1 + (long long)blockIdx.z * sB;
    const float* B2b = B2 + (long long)blockIdx.z * sB;
    float*       Cb  = C  + (long long)blockIdx.z * sC;

    int m0 = blockIdx.y * TM, n0 = blockIdx.x * TN;
    int tid = threadIdx.x, wave = tid >> 5, lane = tid & 31;
    int wm = wave & 3, wn = wave >> 2;
    int l15 = lane & 15;
    int akb = (lane >> 4) << 3, bkb = (lane >> 4) << 3;

    const float* a1pt[4]; const float* a2pt[4]; float amk[4]; bf16_t* a1st[4]; bf16_t* a2st[4];
    #pragma unroll
    for (int j = 0; j < 4; ++j) {
        int i = tid + j * WG, r = i >> 3, c4 = (i & 7) << 2;
        int gr = m0 + r;
        amk[j] = (gr < M) ? 1.0f : 0.0f;
        long long ro = (long long)((gr < M) ? gr : 0) * Kd + c4;
        a1pt[j] = A1 + ro;
        a2pt[j] = A2 + ro;
        a1st[j] = &A1sm[r][c4];
        a2st[j] = &A2sm[r][c4];
    }
    const float* b1pt[2]; const float* b2pt[2]; bf16_t* b1st[2]; bf16_t* b2st[2];
    #pragma unroll
    for (int j = 0; j < 2; ++j) {
        int i = tid + j * WG, r = i >> 4, c4 = (i & 15) << 2;
        b1pt[j] = B1b + (long long)r * N + (n0 + c4);
        b2pt[j] = B2b + (long long)r * N + (n0 + c4);
        b1st[j] = &B1sm[c4][r];
        b2st[j] = &B2sm[c4][r];
    }
    const bf16_t* a1r0 = &A1sm[wm * 32 + l15][0];
    const bf16_t* a1r1 = &A1sm[wm * 32 + 16 + l15][0];
    const bf16_t* a2r0 = &A2sm[wm * 32 + l15][0];
    const bf16_t* a2r1 = &A2sm[wm * 32 + 16 + l15][0];
    const bf16_t* p10 = &B1sm[wn * 32 + l15][bkb];
    const bf16_t* p11 = &B1sm[wn * 32 + 16 + l15][bkb];
    const bf16_t* p20 = &B2sm[wn * 32 + l15][bkb];
    const bf16_t* p21 = &B2sm[wn * 32 + 16 + l15][bkb];

    v8f c00 = {}, c01 = {}, c10 = {}, c11 = {};

    for (int k0 = 0; k0 < Kd; k0 += TK) {
        f32x4 a1v[4], a2v[4], b1v[2], b2v[2];
        #pragma unroll
        for (int j = 0; j < 4; ++j) {
            a1v[j] = *(const f32x4*)a1pt[j]; a1pt[j] += TK;
            a2v[j] = *(const f32x4*)a2pt[j]; a2pt[j] += TK;
        }
        #pragma unroll
        for (int j = 0; j < 2; ++j) {
            b1v[j] = *(const f32x4*)b1pt[j]; b1pt[j] += (long long)TK * N;
            b2v[j] = *(const f32x4*)b2pt[j]; b2pt[j] += (long long)TK * N;
        }
        if (k0 + TK < Kd) __builtin_prefetch(a1pt[0], 0, 1);
        __syncthreads();
        #pragma unroll
        for (int j = 0; j < 4; ++j) {
            float m1 = amk[j], m2 = amk[j] * sgn;
            v4bf h1, h2;
            h1[0] = (bf16_t)(a1v[j][0] * m1); h1[1] = (bf16_t)(a1v[j][1] * m1);
            h1[2] = (bf16_t)(a1v[j][2] * m1); h1[3] = (bf16_t)(a1v[j][3] * m1);
            h2[0] = (bf16_t)(a2v[j][0] * m2); h2[1] = (bf16_t)(a2v[j][1] * m2);
            h2[2] = (bf16_t)(a2v[j][2] * m2); h2[3] = (bf16_t)(a2v[j][3] * m2);
            *(v4bf*)a1st[j] = h1;
            *(v4bf*)a2st[j] = h2;
        }
        #pragma unroll
        for (int j = 0; j < 2; ++j) {
            b1st[j][0 * RSTR] = (bf16_t)b1v[j][0]; b1st[j][1 * RSTR] = (bf16_t)b1v[j][1];
            b1st[j][2 * RSTR] = (bf16_t)b1v[j][2]; b1st[j][3 * RSTR] = (bf16_t)b1v[j][3];
            b2st[j][0 * RSTR] = (bf16_t)b2v[j][0]; b2st[j][1 * RSTR] = (bf16_t)b2v[j][1];
            b2st[j][2 * RSTR] = (bf16_t)b2v[j][2]; b2st[j][3 * RSTR] = (bf16_t)b2v[j][3];
        }
        __syncthreads();

        v16bf a10 = SHUF16(*(const v8bf*)(a1r0 + akb), *(const v8bf*)(a1r0 + 16 + akb));
        v16bf a11 = SHUF16(*(const v8bf*)(a1r1 + akb), *(const v8bf*)(a1r1 + 16 + akb));
        v16bf u0  = SHUF16(*(const v8bf*)(p10), *(const v8bf*)(p10 + 16));
        v16bf u1  = SHUF16(*(const v8bf*)(p11), *(const v8bf*)(p11 + 16));
        c00 = WMMA_BF16(a10, u0, c00); c01 = WMMA_BF16(a10, u1, c01);
        c10 = WMMA_BF16(a11, u0, c10); c11 = WMMA_BF16(a11, u1, c11);

        v16bf a20 = SHUF16(*(const v8bf*)(a2r0 + akb), *(const v8bf*)(a2r0 + 16 + akb));
        v16bf a21 = SHUF16(*(const v8bf*)(a2r1 + akb), *(const v8bf*)(a2r1 + 16 + akb));
        v16bf w0  = SHUF16(*(const v8bf*)(p20), *(const v8bf*)(p20 + 16));
        v16bf w1  = SHUF16(*(const v8bf*)(p21), *(const v8bf*)(p21 + 16));
        c00 = WMMA_BF16(a20, w0, c00); c01 = WMMA_BF16(a20, w1, c01);
        c10 = WMMA_BF16(a21, w0, c10); c11 = WMMA_BF16(a21, w1, c11);
    }

    int rb = (lane >> 4) << 3;
    auto store_acc = [&](v8f& acc, int roff, int coff) {
        int col = n0 + wn * 32 + coff + l15;
        #pragma unroll
        for (int r = 0; r < 8; ++r) {
            int row = m0 + wm * 32 + roff + rb + r;
            if (row < M) Cb[(long long)row * N + col] = acc[r];
        }
    };
    store_acc(c00, 0, 0);  store_acc(c01, 0, 16);
    store_acc(c10, 16, 0); store_acc(c11, 16, 16);
}

// ---------------------------------------------------------------------------
// X = LayerNorm(Y + R) * g + b   (E = 256, one wave per row, 8 contiguous/lane)
// ---------------------------------------------------------------------------
__global__ __launch_bounds__(WG)
void add_ln_kernel(const float* __restrict__ Y, const float* __restrict__ R,
                   float* __restrict__ X,
                   const float* __restrict__ g, const float* __restrict__ bb,
                   int ntok)
{
    int wave = threadIdx.x >> 5, lane = threadIdx.x & 31;
    int row = blockIdx.x * 8 + wave;
    if (row >= ntok) return;
    long long base = (long long)row * 256 + lane * 8;
    f32x4 y0 = *(const f32x4*)(Y + base);
    f32x4 y1 = *(const f32x4*)(Y + base + 4);
    f32x4 r0 = *(const f32x4*)(R + base);
    f32x4 r1 = *(const f32x4*)(R + base + 4);
    float v[8];
    float s = 0.f;
    #pragma unroll
    for (int i = 0; i < 4; ++i) { v[i] = y0[i] + r0[i]; v[i + 4] = y1[i] + r1[i]; }
    #pragma unroll
    for (int i = 0; i < 8; ++i) s += v[i];
    #pragma unroll
    for (int off = 16; off > 0; off >>= 1) s += __shfl_xor(s, off, 32);
    float mean = s * (1.0f / 256.0f);
    float s2 = 0.f;
    #pragma unroll
    for (int i = 0; i < 8; ++i) { float d = v[i] - mean; s2 += d * d; }
    #pragma unroll
    for (int off = 16; off > 0; off >>= 1) s2 += __shfl_xor(s2, off, 32);
    float rs = rsqrtf(s2 * (1.0f / 256.0f) + 1e-5f);
    int e0 = lane * 8;
    f32x4 o0, o1;
    #pragma unroll
    for (int i = 0; i < 4; ++i) {
        o0[i] = (v[i]     - mean) * rs * g[e0 + i]     + bb[e0 + i];
        o1[i] = (v[i + 4] - mean) * rs * g[e0 + 4 + i] + bb[e0 + 4 + i];
    }
    *(f32x4*)(X + base)     = o0;
    *(f32x4*)(X + base + 4) = o1;
}

// ---------------------------------------------------------------------------
// outn = normalize( mask * (a*base + (1-a)*ctx) ), one wave per row
// ---------------------------------------------------------------------------
__global__ __launch_bounds__(WG)
void mix_norm_kernel(const float* __restrict__ base, const float* __restrict__ ctx,
                     const int* __restrict__ toks, float* __restrict__ outn,
                     const float* __restrict__ alpha_p, int ntok)
{
    int wave = threadIdx.x >> 5, lane = threadIdx.x & 31;
    int row = blockIdx.x * 8 + wave;
    if (row >= ntok) return;
    float a   = alpha_p[0];
    float msk = (toks[row] > 0) ? 1.0f : 0.0f;
    long long o = (long long)row * 256 + lane * 8;
    f32x4 b0 = *(const f32x4*)(base + o);
    f32x4 b1 = *(const f32x4*)(base + o + 4);
    f32x4 c0 = *(const f32x4*)(ctx + o);
    f32x4 c1 = *(const f32x4*)(ctx + o + 4);
    float v[8]; float s2 = 0.f;
    #pragma unroll
    for (int i = 0; i < 4; ++i) {
        v[i]     = (a * b0[i] + (1.0f - a) * c0[i]) * msk;
        v[i + 4] = (a * b1[i] + (1.0f - a) * c1[i]) * msk;
    }
    #pragma unroll
    for (int i = 0; i < 8; ++i) s2 += v[i] * v[i];
    #pragma unroll
    for (int off = 16; off > 0; off >>= 1) s2 += __shfl_xor(s2, off, 32);
    float inv = 1.0f / (sqrtf(s2) + 1e-13f);
    f32x4 o0, o1;
    #pragma unroll
    for (int i = 0; i < 4; ++i) { o0[i] = v[i] * inv; o1[i] = v[i + 4] * inv; }
    *(f32x4*)(outn + o)     = o0;
    *(f32x4*)(outn + o + 4) = o1;
}

// ---------------------------------------------------------------------------
// Fused cosine-matrix + tanh + Gaussian kernel pooling.
// grid = (D/64, B). WG: 32 query rows x 64 doc cols of one batch.
// 8 waves: 2 along Q x 4 along D, each one 16x16 WMMA acc over K=256.
// Pooling reduced with 16-lane shfl butterflies -> conflict-free ds_add_f32.
// ---------------------------------------------------------------------------
__global__ __launch_bounds__(WG)
void cos_pool_kernel(const float* __restrict__ qn, const float* __restrict__ dn,
                     const int* __restrict__ qtoks, const int* __restrict__ dtoks,
                     float* __restrict__ summed)
{
    __shared__ alignas(16) bf16_t qs[32][256 + KPAD];
    __shared__ alignas(16) bf16_t dsm[64][256 + KPAD];
    __shared__ float  lsum[32][NKER];
    __shared__ float  qm[32], dm[64];

    int b   = blockIdx.y;
    int d0  = blockIdx.x * 64;
    int tid = threadIdx.x;
    int wave = tid >> 5, lane = tid & 31;
    int wm = wave & 1, wn = wave >> 1;

    const float* qb = qn + (long long)b * 32 * 256;
    const float* db = dn + ((long long)b * 1024 + d0) * 256;

    {   // stage q: gather all loads first, then convert+store
        f32x4 qv[8];
        #pragma unroll
        for (int j = 0; j < 8; ++j) {
            int i = tid + j * WG;
            qv[j] = *(const f32x4*)(qb + (long long)(i >> 6) * 256 + ((i & 63) << 2));
        }
        #pragma unroll
        for (int j = 0; j < 8; ++j) {
            int i = tid + j * WG;
            v4bf h; h[0]=(bf16_t)qv[j][0]; h[1]=(bf16_t)qv[j][1]; h[2]=(bf16_t)qv[j][2]; h[3]=(bf16_t)qv[j][3];
            *(v4bf*)&qs[i >> 6][(i & 63) << 2] = h;
        }
    }
    {   // stage d
        f32x4 dv[16];
        #pragma unroll
        for (int j = 0; j < 16; ++j) {
            int i = tid + j * WG;
            dv[j] = *(const f32x4*)(db + (long long)(i >> 6) * 256 + ((i & 63) << 2));
        }
        #pragma unroll
        for (int j = 0; j < 16; ++j) {
            int i = tid + j * WG;
            v4bf h; h[0]=(bf16_t)dv[j][0]; h[1]=(bf16_t)dv[j][1]; h[2]=(bf16_t)dv[j][2]; h[3]=(bf16_t)dv[j][3];
            *(v4bf*)&dsm[i >> 6][(i & 63) << 2] = h;
        }
    }
    if (tid < 32) qm[tid] = (qtoks[b * 32 + tid] > 0) ? 1.f : 0.f;
    if (tid < 64) dm[tid] = (dtoks[b * 1024 + d0 + tid] > 0) ? 1.f : 0.f;
    for (int i = tid; i < 32 * NKER; i += WG) (&lsum[0][0])[i] = 0.f;
    __syncthreads();

    v8f acc = {};
    int l15 = lane & 15;
    int akb = (lane >> 4) << 3;
    int bkb = (lane >> 4) << 3;
    #pragma unroll
    for (int k0 = 0; k0 < 256; k0 += 32) {
        const bf16_t* ar = &qs[wm * 16 + l15][k0];
        v16bf af = SHUF16(*(const v8bf*)(ar + akb), *(const v8bf*)(ar + 16 + akb));
        const bf16_t* bp = &dsm[wn * 16 + l15][k0 + bkb];
        v16bf bfr = SHUF16(*(const v8bf*)(bp), *(const v8bf*)(bp + 16));
        acc = WMMA_BF16(af, bfr, acc);
    }

    int rb   = (lane >> 4) << 3;
    int dcol = wn * 16 + l15;
    #pragma unroll
    for (int r = 0; r < 8; ++r) {
        int qrow = wm * 16 + rb + r;
        float m = qm[qrow] * dm[dcol];
        float c = tanhf(acc[r] * m);
        #pragma unroll
        for (int kk = 0; kk < NKER; ++kk) {
            float mu  = (kk == 0) ? 1.0f : (0.9f - 0.2f * (float)(kk - 1));
            float i2s = (kk == 0) ? 5.0e7f : 50.0f;       // 1/(2*sigma^2)
            float dd  = c - mu;
            float val = __expf(-dd * dd * i2s) * m;
            // butterfly-sum the 16 doc columns held by this half-wave
            val += __shfl_xor(val, 1, 16);
            val += __shfl_xor(val, 2, 16);
            val += __shfl_xor(val, 4, 16);
            val += __shfl_xor(val, 8, 16);
            if (l15 == 0) atomicAdd(&lsum[qrow][kk], val);
        }
    }
    __syncthreads();
    for (int i = tid; i < 32 * NKER; i += WG)
        atomicAdd(&summed[(long long)b * 32 * NKER + i], (&lsum[0][0])[i]);
}

// ---------------------------------------------------------------------------
// Final score: out[b] = beta*sum_k(log_k*w_slog) + gamma*sum_k(norm_k*w_slen)
// ---------------------------------------------------------------------------
__global__ __launch_bounds__(32)
void score_kernel(const float* __restrict__ summed,
                  const float* __restrict__ wlog, const float* __restrict__ wlen,
                  const float* __restrict__ beta_p, const float* __restrict__ gamma_p,
                  float* __restrict__ out)
{
    int b = blockIdx.x, lane = threadIdx.x;
    const float* sb = summed + (long long)b * 32 * NKER + lane * NKER;
    float sl = 0.f, sn = 0.f;
    #pragma unroll
    for (int kk = 0; kk < NKER; ++kk) {
        float s = sb[kk];
        sl += log2f(fmaxf(s, 1e-10f)) * wlog[kk];
        sn += (s * (1.0f / 1024.0f)) * wlen[kk];
    }
    #pragma unroll
    for (int off = 16; off > 0; off >>= 1) {
        sl += __shfl_xor(sl, off, 32);
        sn += __shfl_xor(sn, off, 32);
    }
    if (lane == 0) out[b] = sl * beta_p[0] + sn * gamma_p[0];
}

// ---------------------------------------------------------------------------
extern "C" void kernel_launch(void* const* d_in, const int* in_sizes, int n_in,
                              void* d_out, int out_size, void* d_ws, size_t ws_size,
                              hipStream_t stream)
{
    (void)in_sizes; (void)n_in; (void)out_size; (void)ws_size;
    const int B = 128, Q = 32, D = 1024, E = 256, L = 2;
    const int NQ = B * Q;
    const long long ND = (long long)B * D;

    const int*   q_toks  = (const int*)d_in[0];
    const int*   d_toks  = (const int*)d_in[1];
    const float* table   = (const float*)d_in[2];
    const float* ln1g    = (const float*)d_in[3];
    const float* ln1b    = (const float*)d_in[4];
    const float* ln2g    = (const float*)d_in[5];
    const float* ln2b    = (const float*)d_in[6];
    const float* W1      = (const float*)d_in[7];
    const float* b1      = (const float*)d_in[8];
    const float* W2      = (const float*)d_in[9];
    const float* b2      = (const float*)d_in[10];
    const float* wlog    = (const float*)d_in[11];
    const float* wlen    = (const float*)d_in[12];
    const float* alpha_p = (const float*)d_in[13];
    const float* beta_p  = (const float*)d_in[14];
    const float* gamma_p = (const float*)d_in[15];
    float* out = (float*)d_out;

    float* ws = (float*)d_ws;
    size_t off = 0;
    auto alloc = [&](size_t n) { float* p = ws + off; off += n; return p; };

    float* q_emb = alloc((size_t)NQ * E);
    float* d_pos = alloc((size_t)ND * E);
    float* qx    = alloc((size_t)NQ * E);
    float* dx    = alloc((size_t)ND * E);
    float* t1    = alloc((size_t)ND * E);
    float* t2    = alloc((size_t)ND * E);
    float* yb    = alloc((size_t)ND * E);     // doc DFT combine, later reused as dn
    float* qn    = alloc((size_t)NQ * E);
    float* qt1   = alloc((size_t)NQ * E);
    float* qt2   = alloc((size_t)NQ * E);
    float* qy    = alloc((size_t)NQ * E);
    float* dftDc = alloc((size_t)1024 * 1024);
    float* dftDs = alloc((size_t)1024 * 1024);
    float* dftEc = alloc((size_t)256 * 256);
    float* dftEs = alloc((size_t)256 * 256);
    float* dftQc = alloc((size_t)32 * 32);
    float* dftQs = alloc((size_t)32 * 32);
    float* summed = alloc((size_t)B * Q * NKER);

    // DFT matrices
    {
        long long t = 1024LL * 1024;
        dft_gen_kernel<<<(int)((t + WG - 1) / WG), WG, 0, stream>>>(dftDc, dftDs, 1024);
        t = 256LL * 256;
        dft_gen_kernel<<<(int)((t + WG - 1) / WG), WG, 0, stream>>>(dftEc, dftEs, 256);
        t = 32LL * 32;
        dft_gen_kernel<<<(int)((t + WG - 1) / WG), WG, 0, stream>>>(dftQc, dftQs, 32);
    }

    // Embeddings + positions (q: raw emb kept + pos->qx; d: pos kept + pos->dx)
    embed_pos_kernel<<<NQ, WG, 0, stream>>>(q_toks, table, q_emb, qx, nullptr, Q);
    embed_pos_kernel<<<(int)ND, WG, 0, stream>>>(d_toks, table, nullptr, d_pos, dx, D);

    const long long sQ = (long long)Q * E;
    const long long sD = (long long)D * E;

    auto gemm = [&](const float* A, const float* Bm, float* C, int M, const float* bias) {
        dim3 g(E / TN, (M + TM - 1) / TM, 1);
        gemm_wmma_kernel<<<g, dim3(WG), 0, stream>>>(A, Bm, C, M, E, E, bias);
    };

    for (int l = 0; l < L; ++l) {
        // ---- query stream ----
        gemm2b_wmma_kernel<<<dim3(E / TN, (NQ + TM - 1) / TM, 1), dim3(WG), 0, stream>>>(
            qx, dftEc, dftEs, qt1, qt2, NQ, E, E);                       // T1,T2 = X @ {cosE,sinE}
        gemm2a_wmma_kernel<<<dim3(E / TN, (Q + TM - 1) / TM, B), dim3(WG), 0, stream>>>(
            dftQc, dftQs, qt1, qt2, qy, Q, E, Q, sQ, sQ, -1.0f);         // Y = cosL@T1 - sinL@T2
        add_ln_kernel<<<NQ / 8, WG, 0, stream>>>(qy, qx, qx, ln1g + l * E, ln1b + l * E, NQ);
        gemm(qx, W1 + (size_t)l * E * E, qt1, NQ, b1 + l * E);
        gemm(qt1, W2 + (size_t)l * E * E, qt2, NQ, b2 + l * E);
        add_ln_kernel<<<NQ / 8, WG, 0, stream>>>(qt2, qx, qx, ln2g + l * E, ln2b + l * E, NQ);

        // ---- document stream ----
        gemm2b_wmma_kernel<<<dim3(E / TN, (int)((ND + TM - 1) / TM), 1), dim3(WG), 0, stream>>>(
            dx, dftEc, dftEs, t1, t2, (int)ND, E, E);
        gemm2a_wmma_kernel<<<dim3(E / TN, (D + TM - 1) / TM, B), dim3(WG), 0, stream>>>(
            dftDc, dftDs, t1, t2, yb, D, E, D, sD, sD, -1.0f);
        add_ln_kernel<<<(int)(ND / 8), WG, 0, stream>>>(yb, dx, dx, ln1g + l * E, ln1b + l * E, (int)ND);
        gemm(dx, W1 + (size_t)l * E * E, t1, (int)ND, b1 + l * E);
        gemm(t1, W2 + (size_t)l * E * E, t2, (int)ND, b2 + l * E);
        add_ln_kernel<<<(int)(ND / 8), WG, 0, stream>>>(t2, dx, dx, ln2g + l * E, ln2b + l * E, (int)ND);
    }

    // Mixing + normalization (q uses raw embeddings, d uses positional — per reference)
    mix_norm_kernel<<<NQ / 8, WG, 0, stream>>>(q_emb, qx, q_toks, qn, alpha_p, NQ);
    mix_norm_kernel<<<(int)(ND / 8), WG, 0, stream>>>(d_pos, dx, d_toks, yb, alpha_p, (int)ND);

    hipMemsetAsync(summed, 0, (size_t)B * Q * NKER * sizeof(float), stream);
    cos_pool_kernel<<<dim3(D / 64, B), WG, 0, stream>>>(qn, yb, q_toks, d_toks, summed);
    score_kernel<<<B, 32, 0, stream>>>(summed, wlog, wlen, beta_p, gamma_p, out);
}
